// LatentRepeat_34428457845274
// MI455X (gfx1250) — compile-verified, matrix operand order checked by reading, not run on hardware
//
#include <hip/hip_runtime.h>
#include <hip/hip_bf16.h>
#include <math.h>

typedef _Float16 half_t;
typedef __attribute__((ext_vector_type(16))) _Float16 v16h;
typedef __attribute__((ext_vector_type(8)))  float    v8f;

#define BDIM 32
#define NTOK 577
#define DDIM 384
#define HDIM 1536
#define LNUM 12
#define NREP 4
#define MMAX 580
#define MPADC 584   // (580+7)&~7 ; constant for all reps
#define EPSF 1e-6f

#define BM 128
#define BN 128
#define BK 32
#define LDSS 40     // 32 + 8 half padding to dodge LDS bank conflicts

// ---------------------------------------------------------------------------
// WMMA GEMM:  C[r,c] = sum_k A[r,k] * W[c,k]   (W is the "B^T" operand, [Nc,K])
// 8 waves; each wave owns a 32x64 sub-tile = 2x4 wmma fragments per K-step.
// Interior tiles are staged with GLOBAL_LOAD_ASYNC_TO_LDS_B128 (ASYNCcnt);
// ragged edges fall back to guarded synchronous staging.
// MODE 0: outF = acc + bias            (bias may be null)
// MODE 1: outF = acc * alpha           (attention logits)
// MODE 2: outH = f16(gelu(acc + bias)) (fc1)
// MODE 3: outF = (acc + bias)*ls + res (proj / fc2 residual, in-place on res)
// ---------------------------------------------------------------------------
template<int MODE>
__global__ __launch_bounds__(256)
void gemm_wmma(const half_t* __restrict__ A, long long strideA, int lda,
               const half_t* __restrict__ Wm, long long strideW, int ldw,
               int Mrows, int Ncols, int K,
               const float* __restrict__ bias,
               const float* __restrict__ ls,
               const float* __restrict__ res,
               float* __restrict__ outF, long long strideOutF, int ldOutF,
               half_t* __restrict__ outH, long long strideOutH, int ldOutH,
               float alpha)
{
    __shared__ half_t As[BM * LDSS];
    __shared__ half_t Bs[BN * LDSS];

    const int bz = blockIdx.z;
    A  += (long long)bz * strideA;
    Wm += (long long)bz * strideW;

    const int tid     = threadIdx.x;
    const int wave    = tid >> 5;
    const int lane    = tid & 31;
    const int half_id = lane >> 4;     // 0/1 : lane halves per CDNA5 WMMA layout
    const int idx     = lane & 15;
    const int wm      = wave & 3;      // 4 waves along M (32 rows each)
    const int wn      = wave >> 2;     // 2 waves along N (64 cols each)

    const int blockRow = blockIdx.y * BM;
    const int blockCol = blockIdx.x * BN;
    const bool fullMN = (blockRow + BM <= Mrows) && (blockCol + BN <= Ncols);

    v8f acc[2][4];
#pragma unroll
    for (int i = 0; i < 2; ++i)
#pragma unroll
        for (int j = 0; j < 4; ++j)
#pragma unroll
            for (int e = 0; e < 8; ++e) acc[i][j][e] = 0.0f;

    const int r0 = tid >> 2;            // 0..63 : row handled by this thread
    const int kc = (tid & 3) * 8;       // 0/8/16/24 : k-offset (halves)

    for (int kt = 0; kt < K; kt += BK) {
        if (fullMN && (kt + BK <= K)) {
            // ---- async staging: global -> LDS, no VGPR data (ASYNCcnt) ----
#pragma unroll
            for (int p = 0; p < 2; ++p) {
                int r = p * 64 + r0;
                {
                    const half_t* src = A + (long long)(blockRow + r) * lda + kt + kc;
                    unsigned dlds = (unsigned)(uintptr_t)(&As[r * LDSS + kc]);
                    asm volatile("global_load_async_to_lds_b128 %0, %1, off"
                                 :: "v"(dlds), "v"(src) : "memory");
                }
                {
                    const half_t* src = Wm + (long long)(blockCol + r) * ldw + kt + kc;
                    unsigned dlds = (unsigned)(uintptr_t)(&Bs[r * LDSS + kc]);
                    asm volatile("global_load_async_to_lds_b128 %0, %1, off"
                                 :: "v"(dlds), "v"(src) : "memory");
                }
            }
            asm volatile("s_wait_asynccnt 0x0" ::: "memory");
        } else {
            // ---- guarded synchronous staging for ragged edges ----
#pragma unroll
            for (int p = 0; p < 2; ++p) {
                int r  = p * 64 + r0;
                int gr = blockRow + r;
                half_t* dst = &As[r * LDSS + kc];
                if (gr < Mrows && (kt + kc + 8) <= K) {
                    *(uint4*)dst = *(const uint4*)(A + (long long)gr * lda + kt + kc);
                } else {
#pragma unroll
                    for (int j = 0; j < 8; ++j) {
                        int kk = kt + kc + j;
                        dst[j] = (gr < Mrows && kk < K)
                                     ? A[(long long)gr * lda + kk] : (half_t)0.0f;
                    }
                }
                int gn = blockCol + r;
                half_t* dstB = &Bs[r * LDSS + kc];
                if (gn < Ncols && (kt + kc + 8) <= K) {
                    *(uint4*)dstB = *(const uint4*)(Wm + (long long)gn * ldw + kt + kc);
                } else {
#pragma unroll
                    for (int j = 0; j < 8; ++j) {
                        int kk = kt + kc + j;
                        dstB[j] = (gn < Ncols && kk < K)
                                      ? Wm[(long long)gn * ldw + kk] : (half_t)0.0f;
                    }
                }
            }
        }
        __syncthreads();

        // ---- fragments per documented CDNA5 16-bit A/B VGPR layout ----
        v16h afrag[2], bfrag[4];
#pragma unroll
        for (int f = 0; f < 2; ++f) {
            const half_t* ap = &As[(wm * 32 + f * 16 + idx) * LDSS];
            union { v16h v; unsigned u[8]; } ua;
#pragma unroll
            for (int vv = 0; vv < 8; ++vv) {
                int kl = ((vv & 3) << 1) + (half_id << 3) + ((vv >> 2) << 4);
                ua.u[vv] = *(const unsigned*)(ap + kl);
            }
            afrag[f] = ua.v;
        }
#pragma unroll
        for (int f = 0; f < 4; ++f) {
            const half_t* bp = &Bs[(wn * 64 + f * 16 + idx) * LDSS];
            union { v16h v; unsigned u[8]; } ub;
#pragma unroll
            for (int vv = 0; vv < 8; ++vv) {
                int kl = ((vv & 3) << 1) + (half_id << 3) + ((vv >> 2) << 4);
                ub.u[vv] = *(const unsigned*)(bp + kl);
            }
            bfrag[f] = ub.v;
        }

#pragma unroll
        for (int i = 0; i < 2; ++i)
#pragma unroll
            for (int j = 0; j < 4; ++j)
                acc[i][j] = __builtin_amdgcn_wmma_f32_16x16x32_f16(
                    false, afrag[i], false, bfrag[j],
                    (short)0, acc[i][j], false, false);
        __syncthreads();
    }

    // ---- epilogue: C layout lane=half*16+idx; VGPR r -> row r + 8*half ----
#pragma unroll
    for (int j = 0; j < 4; ++j) {
        int cb = blockCol + wn * 64 + j * 16 + idx;
        if (cb >= Ncols) continue;
        float bi = (MODE != 1 && bias) ? bias[cb] : 0.0f;
        float lv = (MODE == 3) ? ls[cb] : 0.0f;
#pragma unroll
        for (int i = 0; i < 2; ++i) {
#pragma unroll
            for (int r = 0; r < 8; ++r) {
                int rr = blockRow + wm * 32 + i * 16 + r + 8 * half_id;
                if (rr >= Mrows) continue;
                float a = acc[i][j][r];
                if (MODE == 0) {
                    outF[(long long)bz * strideOutF + (long long)rr * ldOutF + cb] = a + bi;
                } else if (MODE == 1) {
                    outF[(long long)bz * strideOutF + (long long)rr * ldOutF + cb] = a * alpha;
                } else if (MODE == 2) {
                    float t = a + bi;
                    float g = 0.5f * t * (1.0f + erff(t * 0.70710678118654752f));
                    outH[(long long)bz * strideOutH + (long long)rr * ldOutH + cb] = (half_t)g;
                } else { // MODE 3
                    long long o = (long long)bz * strideOutF + (long long)rr * ldOutF + cb;
                    outF[o] = (a + bi) * lv + res[o];
                }
            }
        }
    }
}

// ---------------------------------------------------------------------------
// LayerNorm over D=384 (f32 in -> f16 out), one wave32 per row
// ---------------------------------------------------------------------------
__global__ __launch_bounds__(256)
void ln_kernel(const float* __restrict__ x, const float* __restrict__ w,
               const float* __restrict__ b, half_t* __restrict__ out,
               long long rows)
{
    int wave = threadIdx.x >> 5;
    int lane = threadIdx.x & 31;
    long long row = (long long)blockIdx.x * 8 + wave;
    if (row >= rows) return;
    const float* xr = x + row * DDIM;
    float v[DDIM / 32];
    float s = 0.0f;
#pragma unroll
    for (int j = 0; j < DDIM / 32; ++j) { v[j] = xr[lane + j * 32]; s += v[j]; }
#pragma unroll
    for (int o = 16; o > 0; o >>= 1) s += __shfl_xor(s, o, 32);
    float mu = s * (1.0f / DDIM);
    float q = 0.0f;
#pragma unroll
    for (int j = 0; j < DDIM / 32; ++j) { float d = v[j] - mu; q += d * d; }
#pragma unroll
    for (int o = 16; o > 0; o >>= 1) q += __shfl_xor(q, o, 32);
    float rstd = rsqrtf(q * (1.0f / DDIM) + EPSF);
    half_t* orow = out + row * DDIM;
#pragma unroll
    for (int j = 0; j < DDIM / 32; ++j) {
        int c = lane + j * 32;
        orow[c] = (half_t)((v[j] - mu) * rstd * w[c] + b[c]);
    }
}

// softmax over row of length Mlen (f32 logits -> f16 probs, pad to Mpad=0)
__global__ __launch_bounds__(256)
void softmax_kernel(const float* __restrict__ S, half_t* __restrict__ P,
                    int Mlen, int Mpad)
{
    long long row = blockIdx.x;
    const float* s = S + row * Mpad;
    half_t* p = P + row * Mpad;
    __shared__ float red[256];
    int tid = threadIdx.x;
    float lmax = -3.0e38f;
    for (int i = tid; i < Mlen; i += 256) lmax = fmaxf(lmax, s[i]);
    red[tid] = lmax; __syncthreads();
    for (int st = 128; st > 0; st >>= 1) {
        if (tid < st) red[tid] = fmaxf(red[tid], red[tid + st]);
        __syncthreads();
    }
    float m = red[0]; __syncthreads();
    float lsum = 0.0f;
    for (int i = tid; i < Mlen; i += 256) lsum += expf(s[i] - m);
    red[tid] = lsum; __syncthreads();
    for (int st = 128; st > 0; st >>= 1) {
        if (tid < st) red[tid] += red[tid + st];
        __syncthreads();
    }
    float inv = 1.0f / red[0];
    for (int i = tid; i < Mpad; i += 256)
        p[i] = (i < Mlen) ? (half_t)(expf(s[i] - m) * inv) : (half_t)0.0f;
}

__global__ void cvt_kernel(const float* __restrict__ in, half_t* __restrict__ out,
                           long long n)
{
    long long i = (long long)blockIdx.x * 256 + threadIdx.x;
    if (i < n) out[i] = (half_t)in[i];
}

// v [B, Mlen, D] f32 -> vT [B, D, Mpad] f16 (pad zeroed)
__global__ void transpose_v_kernel(const float* __restrict__ v32,
                                   half_t* __restrict__ vT, int Mlen, int Mpad)
{
    long long i = (long long)blockIdx.x * 256 + threadIdx.x;
    long long total = (long long)BDIM * DDIM * Mpad;
    if (i >= total) return;
    int m = (int)(i % Mpad);
    long long t = i / Mpad;
    int d = (int)(t % DDIM);
    int b = (int)(t / DDIM);
    vT[i] = (m < Mlen) ? (half_t)v32[((long long)b * Mlen + m) * DDIM + d]
                       : (half_t)0.0f;
}

// rep start: keep x row0, reset rows 1.. from original input, record cls into
// history slot r, and emit the fixed per-rep f16 memory [B, Mlen, D]
__global__ void rep_build_kernel(float* __restrict__ xw,
                                 const float* __restrict__ xp,
                                 float* __restrict__ hist,
                                 half_t* __restrict__ mem16, int r, int Mlen)
{
    long long i = (long long)blockIdx.x * 256 + threadIdx.x;
    long long total = (long long)BDIM * Mlen * DDIM;
    if (i >= total) return;
    int d = (int)(i % DDIM);
    long long t = i / DDIM;
    int row = (int)(t % Mlen);
    int b = (int)(t / Mlen);
    float val;
    if (row == 0) {
        val = xw[((long long)b * NTOK) * DDIM + d];
        hist[((long long)r * BDIM + b) * DDIM + d] = val;
    } else if (row < NTOK) {
        val = xp[((long long)b * NTOK + row) * DDIM + d];
        xw[((long long)b * NTOK + row) * DDIM + d] = val;
    } else {
        int j = row - NTOK;
        val = hist[((long long)j * BDIM + b) * DDIM + d];
    }
    mem16[i] = (half_t)val;
}

// ---------------------------------------------------------------------------
static void launch_gemm(int mode, hipStream_t stream,
                        const half_t* A, long long sA, int lda,
                        const half_t* W, long long sW, int ldw,
                        int Mr, int Nc, int K, int batch,
                        const float* bias, const float* ls, const float* res,
                        float* oF, long long sOF, int ldOF,
                        half_t* oH, long long sOH, int ldOH, float alpha)
{
    dim3 grid((Nc + BN - 1) / BN, (Mr + BM - 1) / BM, batch);
    dim3 block(256);
    switch (mode) {
    case 0: gemm_wmma<0><<<grid, block, 0, stream>>>(A, sA, lda, W, sW, ldw, Mr, Nc, K, bias, ls, res, oF, sOF, ldOF, oH, sOH, ldOH, alpha); break;
    case 1: gemm_wmma<1><<<grid, block, 0, stream>>>(A, sA, lda, W, sW, ldw, Mr, Nc, K, bias, ls, res, oF, sOF, ldOF, oH, sOH, ldOH, alpha); break;
    case 2: gemm_wmma<2><<<grid, block, 0, stream>>>(A, sA, lda, W, sW, ldw, Mr, Nc, K, bias, ls, res, oF, sOF, ldOF, oH, sOH, ldOH, alpha); break;
    default: gemm_wmma<3><<<grid, block, 0, stream>>>(A, sA, lda, W, sW, ldw, Mr, Nc, K, bias, ls, res, oF, sOF, ldOF, oH, sOH, ldOH, alpha); break;
    }
}

extern "C" void kernel_launch(void* const* d_in, const int* in_sizes, int n_in,
                              void* d_out, int out_size, void* d_ws, size_t ws_size,
                              hipStream_t stream)
{
    (void)in_sizes; (void)n_in; (void)out_size; (void)ws_size;
    const float* in_x  = (const float*)d_in[0];
    const float* n1w   = (const float*)d_in[1];
    const float* n1b   = (const float*)d_in[2];
    const float* qkvw  = (const float*)d_in[3];
    const float* qkvb  = (const float*)d_in[4];
    const float* qnw   = (const float*)d_in[5];
    const float* qnb   = (const float*)d_in[6];
    const float* knw   = (const float*)d_in[7];
    const float* knb   = (const float*)d_in[8];
    const float* anw   = (const float*)d_in[9];
    const float* anb   = (const float*)d_in[10];
    const float* projw = (const float*)d_in[11];
    const float* projb = (const float*)d_in[12];
    const float* ls1   = (const float*)d_in[13];
    const float* n2w   = (const float*)d_in[14];
    const float* n2b   = (const float*)d_in[15];
    const float* fc1w  = (const float*)d_in[16];
    const float* fc1b  = (const float*)d_in[17];
    const float* fc2w  = (const float*)d_in[18];
    const float* fc2b  = (const float*)d_in[19];
    const float* ls2   = (const float*)d_in[20];

    const long long XE = (long long)BDIM * NTOK * DDIM;      // 7,090,176
    const long long MEm = (long long)BDIM * MMAX * DDIM;     // 7,127,040
    const long long HE = (long long)BDIM * NTOK * HDIM;      // 28,360,704
    const long long SEp = (long long)BDIM * NTOK * MPADC;    // logits (padded)

    char* ws = (char*)d_ws;
    size_t off = 0;
    auto alloc = [&](size_t bytes) {
        off = (off + 255) & ~(size_t)255;
        size_t o = off; off += bytes; return o;
    };
    float*  xw     = (float*)(ws + alloc(XE * 4));
    float*  hist   = (float*)(ws + alloc((size_t)NREP * BDIM * DDIM * 4));
    half_t* mem16  = (half_t*)(ws + alloc(MEm * 2));
    half_t* act16  = (half_t*)(ws + alloc(XE * 2));
    float*  tmpA   = (float*)(ws + alloc(XE * 4));           // q f32, then attn-out f32
    float*  k32    = (float*)(ws + alloc(MEm * 4));
    float*  v32    = (float*)(ws + alloc(MEm * 4));
    half_t* q16    = (half_t*)(ws + alloc(XE * 2));
    half_t* k16    = (half_t*)(ws + alloc(MEm * 2));
    half_t* vT16   = (half_t*)(ws + alloc((size_t)BDIM * DDIM * MPADC * 2));
    size_t arenaB  = SEp * 6 > HE * 2 ? (size_t)(SEp * 6) : (size_t)(HE * 2);
    char*   arena  = ws + alloc(arenaB);
    float*  scores = (float*)arena;
    half_t* probs  = (half_t*)(arena + SEp * 4);
    half_t* h16    = (half_t*)arena;                         // reuses attn arena
    half_t* qkv16  = (half_t*)(ws + alloc((size_t)LNUM * 3 * DDIM * DDIM * 2));
    half_t* proj16 = (half_t*)(ws + alloc((size_t)LNUM * DDIM * DDIM * 2));
    half_t* fc116  = (half_t*)(ws + alloc((size_t)LNUM * HDIM * DDIM * 2));
    half_t* fc216  = (half_t*)(ws + alloc((size_t)LNUM * DDIM * HDIM * 2));

    // residual stream starts as the input
    hipMemcpyAsync(xw, in_x, XE * 4, hipMemcpyDeviceToDevice, stream);

    // weights -> f16
    auto cvt = [&](const float* s, half_t* d, long long n) {
        cvt_kernel<<<(unsigned)((n + 255) / 256), 256, 0, stream>>>(s, d, n);
    };
    cvt(qkvw, qkv16, (long long)LNUM * 3 * DDIM * DDIM);
    cvt(projw, proj16, (long long)LNUM * DDIM * DDIM);
    cvt(fc1w, fc116, (long long)LNUM * HDIM * DDIM);
    cvt(fc2w, fc216, (long long)LNUM * DDIM * HDIM);

    const float scale = 1.0f / sqrtf((float)DDIM);
    const long long rowsX = (long long)BDIM * NTOK;

    for (int r = 0; r < NREP; ++r) {
        const int Mlen = NTOK + r;
        const long long rowsM = (long long)BDIM * Mlen;

        rep_build_kernel<<<(unsigned)((rowsM * DDIM + 255) / 256), 256, 0, stream>>>(
            xw, in_x, hist, mem16, r, Mlen);

        for (int l = 0; l < LNUM; ++l) {
            const half_t* Wq = qkv16 + (long long)l * 3 * DDIM * DDIM;
            const half_t* Wk = Wq + (long long)DDIM * DDIM;
            const half_t* Wv = Wk + (long long)DDIM * DDIM;
            const float* bq = qkvb + (long long)l * 3 * DDIM;
            const float* bk = bq + DDIM;
            const float* bv = bk + DDIM;

            // norm1 -> act16
            ln_kernel<<<(unsigned)((rowsX + 7) / 8), 256, 0, stream>>>(
                xw, n1w + l * DDIM, n1b + l * DDIM, act16, rowsX);
            // q = xn @ Wq^T + bq   (f32 -> tmpA)
            launch_gemm(0, stream, act16, 0, DDIM, Wq, 0, DDIM,
                        (int)rowsX, DDIM, DDIM, 1, bq, nullptr, nullptr,
                        tmpA, 0, DDIM, nullptr, 0, 0, 0.0f);
            // k = mem @ Wk^T + bk ; v = mem @ Wv^T + bv
            launch_gemm(0, stream, mem16, 0, DDIM, Wk, 0, DDIM,
                        (int)rowsM, DDIM, DDIM, 1, bk, nullptr, nullptr,
                        k32, 0, DDIM, nullptr, 0, 0, 0.0f);
            launch_gemm(0, stream, mem16, 0, DDIM, Wv, 0, DDIM,
                        (int)rowsM, DDIM, DDIM, 1, bv, nullptr, nullptr,
                        v32, 0, DDIM, nullptr, 0, 0, 0.0f);
            // q/k layernorms -> f16
            ln_kernel<<<(unsigned)((rowsX + 7) / 8), 256, 0, stream>>>(
                tmpA, qnw + l * DDIM, qnb + l * DDIM, q16, rowsX);
            ln_kernel<<<(unsigned)((rowsM + 7) / 8), 256, 0, stream>>>(
                k32, knw + l * DDIM, knb + l * DDIM, k16, rowsM);
            // v -> f16 transposed [B, D, Mpad]
            transpose_v_kernel<<<(unsigned)(((long long)BDIM * DDIM * MPADC + 255) / 256),
                                 256, 0, stream>>>(v32, vT16, Mlen, MPADC);
            // logits = scale * q @ k^T   (batched over B)
            launch_gemm(1, stream, q16, (long long)NTOK * DDIM, DDIM,
                        k16, (long long)Mlen * DDIM, DDIM,
                        NTOK, Mlen, DDIM, BDIM, nullptr, nullptr, nullptr,
                        scores, (long long)NTOK * MPADC, MPADC,
                        nullptr, 0, 0, scale);
            softmax_kernel<<<(unsigned)rowsX, 256, 0, stream>>>(scores, probs, Mlen, MPADC);
            // attn @ v -> tmpA (f32)
            launch_gemm(0, stream, probs, (long long)NTOK * MPADC, MPADC,
                        vT16, (long long)DDIM * MPADC, MPADC,
                        NTOK, DDIM, Mlen, BDIM, nullptr, nullptr, nullptr,
                        tmpA, (long long)NTOK * DDIM, DDIM, nullptr, 0, 0, 0.0f);
            // attn-norm -> act16
            ln_kernel<<<(unsigned)((rowsX + 7) / 8), 256, 0, stream>>>(
                tmpA, anw + l * DDIM, anb + l * DDIM, act16, rowsX);
            // x = (proj(prod) + b)*ls1 + x
            launch_gemm(3, stream, act16, 0, DDIM,
                        proj16 + (long long)l * DDIM * DDIM, 0, DDIM,
                        (int)rowsX, DDIM, DDIM, 1,
                        projb + l * DDIM, ls1 + l * DDIM, xw,
                        xw, 0, DDIM, nullptr, 0, 0, 0.0f);
            // norm2 -> act16
            ln_kernel<<<(unsigned)((rowsX + 7) / 8), 256, 0, stream>>>(
                xw, n2w + l * DDIM, n2b + l * DDIM, act16, rowsX);
            // h = gelu(fc1) -> f16
            launch_gemm(2, stream, act16, 0, DDIM,
                        fc116 + (long long)l * HDIM * DDIM, 0, DDIM,
                        (int)rowsX, HDIM, DDIM, 1,
                        fc1b + l * HDIM, nullptr, nullptr,
                        nullptr, 0, 0, h16, 0, HDIM, 0.0f);
            // x = (fc2(h) + b)*ls2 + x
            launch_gemm(3, stream, h16, 0, HDIM,
                        fc216 + (long long)l * DDIM * HDIM, 0, HDIM,
                        (int)rowsX, DDIM, HDIM, 1,
                        fc2b + l * DDIM, ls2 + l * DDIM, xw,
                        xw, 0, DDIM, nullptr, 0, 0, 0.0f);
        }
    }

    hipMemcpyAsync(d_out, xw, XE * 4, hipMemcpyDeviceToDevice, stream);
}
